// ConvexLayer4_60894046323183
// MI455X (gfx1250) — matrix operand order repaired and checked by reading, not compiled, not run on hardware
//
#include <hip/hip_runtime.h>

typedef _Float16 v16h __attribute__((ext_vector_type(16)));
typedef float    v8f  __attribute__((ext_vector_type(8)));

// out[b,f] = used(x) * ( b2[f] + sum_hh sigmoid(na(x)*W1[f,0,hh] + k[f]*W1[f,1,hh] + b1[f,hh]) * W2[f,hh] )
// na(x) = ln(0.1/x), NaN->0.  noise / h_higher / h_lower are dead code for the returned output.
__global__ __launch_bounds__(256)
void convex_mlp_wmma_kernel(const float* __restrict__ inputs,
                            const float* __restrict__ kvec,
                            const float* __restrict__ W1,
                            const float* __restrict__ b1,
                            const float* __restrict__ W2,
                            const float* __restrict__ b2,
                            float* __restrict__ out)
{
    constexpr int   F      = 256;
    constexpr float NLOG2E = -1.4426950408889634f;  // -log2(e), folded into weights
    constexpr float LN2    = 0.6931471805599453f;
    constexpr float LN01   = -2.302585092994046f;   // ln(0.1)

    // per-block f-tile weights (16 features x 32 hidden)
    __shared__ __align__(32) float    s_w1a[16 * 32];   // -log2e * W1[f,0,hh]
    __shared__ __align__(32) float    s_c  [16 * 32];   // -log2e * (k[f]*W1[f,1,hh] + b1[f,hh])
    __shared__ __align__(32) _Float16 s_w2 [16 * 32];   // W2[f,hh] as f16 (WMMA B operand)
    // per-block input tile (128 b x 16 f), stride 17 to avoid bank conflicts
    __shared__ float s_na[128 * 17];
    __shared__ float s_x [128 * 17];

    const int tid = threadIdx.x;
    const int f0  = blockIdx.x * 16;   // feature tile base
    const int b0  = blockIdx.y * 128;  // batch tile base

    // stage weights: 512 elements / 256 threads
    for (int i = tid; i < 16 * 32; i += 256) {
        const int f  = i >> 5;
        const int hh = i & 31;
        const int gf = f0 + f;
        const float w1a = W1[(size_t)(gf * 2 + 0) * 32 + hh];
        const float w1b = W1[(size_t)(gf * 2 + 1) * 32 + hh];
        s_w1a[i] = NLOG2E * w1a;
        s_c[i]   = NLOG2E * fmaf(kvec[gf], w1b, b1[(size_t)gf * 32 + hh]);
        s_w2[i]  = (_Float16)W2[(size_t)gf * 32 + hh];
    }
    // stage inputs + na: 2048 elements / 256 threads
    for (int i = tid; i < 128 * 16; i += 256) {
        const int bb = i >> 4;
        const int f  = i & 15;
        const float x  = inputs[(size_t)(b0 + bb) * F + f0 + f];
        float na = LN01 - LN2 * __builtin_amdgcn_logf(x);  // ln(0.1) - ln(x); NaN for x<0
        na = (na != na) ? 0.0f : na;                       // log_filter_nan
        s_x [bb * 17 + f] = x;
        s_na[bb * 17 + f] = na;
    }
    __syncthreads();

    const int lane = tid & 31;
    const int wave = tid >> 5;
    const int hb   = lane >> 4;   // half-wave id (K-group select per ISA A/B layouts)
    const int bl   = lane & 15;   // row within A (b), also N column at store
    const int wb   = wave << 4;   // wave's 16-row batch offset within block

    // B operand is loop-invariant per lane: lane contributes only column N==bl,
    // whose values are W2[bl][K], K = 16*hb + e (contiguous 16 halves). Load once.
    const v16h myw2 = *(const v16h*)&s_w2[bl * 32 + (hb << 4)];
    const v16h zz   = {};

    v8f acc = {};

    // Block-diagonal GEMM: 16 x v_wmma_f32_16x16x32_f16 accumulate the 16b x 16f tile.
    for (int fl = 0; fl < 16; ++fl) {
        const float na = s_na[(wb + bl) * 17 + fl];
        const float* wp = &s_w1a[fl * 32];
        const float* cp = &s_c  [fl * 32];
        // A-operand K-groups for this half-wave: K in {16*hb..16*hb+7} and {16*hb+8..16*hb+15}
        const v8f wlo = *(const v8f*)&wp[hb * 16];
        const v8f whi = *(const v8f*)&wp[hb * 16 + 8];
        const v8f clo = *(const v8f*)&cp[hb * 16];
        const v8f chi = *(const v8f*)&cp[hb * 16 + 8];

        v16h a;
        #pragma unroll
        for (int e = 0; e < 8; ++e) {
            const float x2 = fmaf(na, wlo[e], clo[e]);                       // -log2e*(na*w+c)
            const float s0 = __builtin_amdgcn_rcpf(1.0f + __builtin_amdgcn_exp2f(x2));
            a[e] = (_Float16)s0;
            const float x3 = fmaf(na, whi[e], chi[e]);
            const float s1 = __builtin_amdgcn_rcpf(1.0f + __builtin_amdgcn_exp2f(x3));
            a[e + 8] = (_Float16)s1;
        }

        // B-operand: column fl = W2[fl, :], other columns zero.
        const v16h bm = (bl == fl) ? myw2 : zz;

        acc = __builtin_amdgcn_wmma_f32_16x16x32_f16(
            /*neg_a=*/false, a, /*neg_b=*/false, bm,
            /*c_mod=*/(short)0, acc, /*reuse_a=*/false, /*reuse_b=*/false);
    }

    // C/D f32 layout: VGPR v -> M = v + 8*hb, N = lane%16
    const float bias = b2[f0 + bl];
    #pragma unroll
    for (int v = 0; v < 8; ++v) {
        const int M  = v + (hb << 3);
        const int bb = wb + M;
        const float x    = s_x[bb * 17 + bl];
        const float used = (x > -0.5f) ? 1.0f : 0.0f;
        out[(size_t)(b0 + bb) * F + f0 + bl] = (acc[v] + bias) * used;
    }
}

extern "C" void kernel_launch(void* const* d_in, const int* in_sizes, int n_in,
                              void* d_out, int out_size, void* d_ws, size_t ws_size,
                              hipStream_t stream) {
    (void)in_sizes; (void)n_in; (void)out_size; (void)d_ws; (void)ws_size;
    // setup_inputs order: inputs, noise(unused), k, W1, b1, W2, b2
    const float* inputs = (const float*)d_in[0];
    const float* kvec   = (const float*)d_in[2];
    const float* W1     = (const float*)d_in[3];
    const float* b1v    = (const float*)d_in[4];
    const float* W2     = (const float*)d_in[5];
    const float* b2v    = (const float*)d_in[6];
    float* out = (float*)d_out;

    dim3 grid(256 / 16, 4096 / 128);  // (f-tiles, b-tiles) = (16, 32)
    convex_mlp_wmma_kernel<<<grid, dim3(256), 0, stream>>>(inputs, kvec, W1, b1v, W2, b2v, out);
}
